// NodeAlignNodeLossAttention_28501402976840
// MI455X (gfx1250) — matrix operand order repaired and checked by reading, not compiled, not run on hardware
//
#include <hip/hip_runtime.h>

// ---------------------------------------------------------------------------
// CDNA5 / gfx1250 implementation of NodeAlignNodeLoss attention pipeline.
// Dense GEMMs: v_wmma_f32_16x16x32_f16 (f16 in, f32 accum).
// Edge-message MLP: 64 edges/workgroup, B-fragment reuse across 4 M-tiles,
// LDS-staged ReLU intermediate. Pair-score kernel stages LDS tiles with
// global_load_async_to_lds_b128 (ASYNCcnt) per ISA 15.18.3.
// ---------------------------------------------------------------------------

typedef __attribute__((ext_vector_type(16))) _Float16 v16h;
typedef __attribute__((ext_vector_type(8)))  _Float16 v8h;
typedef __attribute__((ext_vector_type(8)))  float    v8f;

__device__ __forceinline__ v8f zero8() {
  v8f z;
#pragma unroll
  for (int i = 0; i < 8; ++i) z[i] = 0.0f;
  return z;
}

__device__ __forceinline__ v8f wmma16(v16h a, v16h b, v8f c) {
  // (neg_a, A, neg_b, B, c_mod, C, reuse_a, reuse_b)
  return __builtin_amdgcn_wmma_f32_16x16x32_f16(false, a, false, b, (short)0, c,
                                                false, false);
}

__device__ __forceinline__ v16h combine8(v8h c0, v8h c1) {
  v16h a;
#pragma unroll
  for (int i = 0; i < 8; ++i) { a[i] = c0[i]; a[i + 8] = c1[i]; }
  return a;
}

// A-fragment (16x32 f16): lane holds row M = lane%16; K chunks at 8*hs and
// 16+8*hs (ISA 7.12.2). rowp points at row start + k-tile base.
__device__ __forceinline__ v16h a_row(const _Float16* rowp, int hs) {
  v8h c0 = *(const v8h*)(rowp + 8 * hs);
  v8h c1 = *(const v8h*)(rowp + 16 + 8 * hs);
  return combine8(c0, c1);
}

__device__ __forceinline__ v8h cvt8f(const float* p) {
  v8h r;
#pragma unroll
  for (int i = 0; i < 8; ++i) r[i] = (_Float16)p[i];
  return r;
}

// B-fragment from pre-packed buffer: one aligned 32B load per lane.
__device__ __forceinline__ v16h b_frag(const _Float16* pw, int tile, int lane) {
  return *(const v16h*)(pw + ((size_t)tile * 32 + lane) * 16);
}

// ---------------------------------------------------------------------------
// Utility kernels
// ---------------------------------------------------------------------------
__global__ void k_zero_u32(unsigned int* p, long n) {
  long i = (long)blockIdx.x * blockDim.x + threadIdx.x;
  if (i < n) p[i] = 0u;
}

__global__ void k_cvt_f16(const float* __restrict__ s, _Float16* __restrict__ d, long n) {
  long i = (long)blockIdx.x * blockDim.x + threadIdx.x;
  if (i < n) d[i] = (_Float16)s[i];
}

// [rows,16] f32 -> [rows,32] f16 with zero pad of K=16..31 (edge features)
__global__ void k_cvt_pad32(const float* __restrict__ s, _Float16* __restrict__ d, long rows) {
  long n = rows * 32;
  long i = (long)blockIdx.x * blockDim.x + threadIdx.x;
  if (i < n) {
    long r = i >> 5;
    int c = (int)(i & 31);
    d[i] = (c < 16) ? (_Float16)s[r * 16 + c] : (_Float16)0.0f;
  }
}

// Pack row-major f32 weight [K,NC] into WMMA B-fragment order (f16), zero
// padded to KPAD. Lane holds col n = lane%16 of its n-tile, halves j=0..15 ->
// K = kt*32 + 8*(lane/16) + j (j<8), kt*32 + 16 + 8*(lane/16) + (j-8) (j>=8).
__global__ void k_pack_bfrag(const float* __restrict__ W, _Float16* __restrict__ out,
                             int K, int KPAD, int NC) {
  int NT = NC >> 4;
  int tile = blockIdx.x;  // tile = kt*NT + nt
  int kt = tile / NT;
  int nt = tile - kt * NT;
  int lane = threadIdx.x;
  int hs = lane >> 4;
  int n = (lane & 15) + nt * 16;
  v16h f;
#pragma unroll
  for (int j = 0; j < 16; ++j) {
    int k = kt * 32 + (j < 8 ? hs * 8 + j : 16 + hs * 8 + (j - 8));
    f[j] = (k < K) ? (_Float16)W[(size_t)k * NC + n] : (_Float16)0.0f;
  }
  *(v16h*)(out + ((size_t)tile * 32 + lane) * 16) = f;
}

// ---------------------------------------------------------------------------
// Encoders
// ---------------------------------------------------------------------------
__global__ void k_encode_node(const _Float16* __restrict__ nf16,
                              const _Float16* __restrict__ pw,
                              const float* __restrict__ bias,
                              float* __restrict__ h, _Float16* __restrict__ h16, int N) {
  int tile = blockIdx.x;
  int lane = threadIdx.x & 31, nt = threadIdx.x >> 5;  // 8 waves = 8 n-tiles
  int hs = lane >> 4, lm = lane & 15;
  int arow = tile * 16 + lm; if (arow >= N) arow = N - 1;
  v16h a = a_row(nf16 + (size_t)arow * 32, hs);
  v16h b = b_frag(pw, nt, lane);
  v8f acc = zero8();
  acc = wmma16(a, b, acc);
  int col = nt * 16 + lm;
  float bv = bias[col];
  int nbase = tile * 16 + 8 * hs;
  if (tile * 16 + 16 <= N) {  // uniform fast path: no per-row exec churn
#pragma unroll
    for (int r = 0; r < 8; ++r) {
      float v = acc[r] + bv;
      size_t ix = (size_t)(nbase + r) * 128 + col;
      h[ix] = v;
      h16[ix] = (_Float16)v;
    }
  } else {
#pragma unroll
    for (int r = 0; r < 8; ++r) {
      int node = nbase + r;
      if (node < N) {
        float v = acc[r] + bv;
        h[(size_t)node * 128 + col] = v;
        h16[(size_t)node * 128 + col] = (_Float16)v;
      }
    }
  }
}

__global__ void k_encode_edge(const _Float16* __restrict__ ef16,
                              const _Float16* __restrict__ pw,
                              const float* __restrict__ bias,
                              _Float16* __restrict__ e16, long E) {
  int tile = blockIdx.x;
  int lane = threadIdx.x & 31, nt = threadIdx.x >> 5;  // 4 waves = 4 n-tiles
  int hs = lane >> 4, lm = lane & 15;
  long arow = (long)tile * 16 + lm; if (arow >= E) arow = E - 1;
  v16h a = a_row(ef16 + (size_t)arow * 32, hs);
  v16h b = b_frag(pw, nt, lane);
  v8f acc = zero8();
  acc = wmma16(a, b, acc);
  int col = nt * 16 + lm;
  float bv = bias[col];
  long ebase = (long)tile * 16 + 8 * hs;
  if ((long)tile * 16 + 16 <= E) {
#pragma unroll
    for (int r = 0; r < 8; ++r)
      e16[(size_t)(ebase + r) * 64 + col] = (_Float16)(acc[r] + bv);
  } else {
#pragma unroll
    for (int r = 0; r < 8; ++r) {
      long em = ebase + r;
      if (em < E) e16[(size_t)em * 64 + col] = (_Float16)(acc[r] + bv);
    }
  }
}

// ---------------------------------------------------------------------------
// Fused 2-layer edge message MLP + segment-sum. 64 edges per workgroup
// (4 M-tiles reuse each B fragment); hidden [64,256] f16 in LDS (32 KB).
// ---------------------------------------------------------------------------
__global__ __launch_bounds__(256) void k_edge_msg(
    const _Float16* __restrict__ h16, const _Float16* __restrict__ e16,
    const int* __restrict__ from_idx, const int* __restrict__ to_idx,
    const _Float16* __restrict__ pw1, const float* __restrict__ b1,
    const _Float16* __restrict__ pw2, const float* __restrict__ b2,
    float* __restrict__ agg, long E) {
  __shared__ __align__(16) _Float16 hid[64 * 256];
  int tile0 = blockIdx.x * 4;
  int lane = threadIdx.x & 31, wv = threadIdx.x >> 5;
  int hs = lane >> 4, lm = lane & 15;

  const _Float16* hf[4];
  const _Float16* ht[4];
  const _Float16* he[4];
#pragma unroll
  for (int m = 0; m < 4; ++m) {
    long em = (long)(tile0 + m) * 16 + lm; if (em >= E) em = E - 1;
    hf[m] = h16 + (size_t)from_idx[em] * 128;
    ht[m] = h16 + (size_t)to_idx[em] * 128;
    he[m] = e16 + (size_t)em * 64;
  }

  // ---- layer 1: edge_in[64,320] @ W_msg1[320,256], relu -> LDS f16 ----
#pragma unroll
  for (int t2 = 0; t2 < 2; ++t2) {
    int nt = wv + t2 * 8;
    v8f acc[4];
#pragma unroll
    for (int m = 0; m < 4; ++m) acc[m] = zero8();
#pragma unroll
    for (int kt = 0; kt < 10; ++kt) {
      int kb = kt * 32;
      int c0 = kb + 8 * hs, c1 = kb + 16 + 8 * hs;
      v16h b = b_frag(pw1, kt * 16 + nt, lane);
      if (kt < 9)  // global_prefetch_b8 of next B tile
        __builtin_prefetch(pw1 + ((size_t)((kt + 1) * 16 + nt) * 32 + lane) * 16, 0, 1);
#pragma unroll
      for (int m = 0; m < 4; ++m) {
        v8h x0 = (c0 < 128) ? *(const v8h*)(hf[m] + c0)
               : (c0 < 256) ? *(const v8h*)(ht[m] + (c0 - 128))
                            : *(const v8h*)(he[m] + (c0 - 256));
        v8h x1 = (c1 < 128) ? *(const v8h*)(hf[m] + c1)
               : (c1 < 256) ? *(const v8h*)(ht[m] + (c1 - 128))
                            : *(const v8h*)(he[m] + (c1 - 256));
        acc[m] = wmma16(combine8(x0, x1), b, acc[m]);
      }
    }
    int col = nt * 16 + lm;
    float bv = b1[col];
#pragma unroll
    for (int m = 0; m < 4; ++m) {
#pragma unroll
      for (int r = 0; r < 8; ++r) {
        float v = acc[m][r] + bv;
        v = v > 0.0f ? v : 0.0f;
        hid[(m * 16 + r + 8 * hs) * 256 + col] = (_Float16)v;
      }
    }
  }
  __syncthreads();

  // ---- layer 2: hidden[64,256] @ W_msg2[256,256] -> atomic segment sum ----
#pragma unroll
  for (int t2 = 0; t2 < 2; ++t2) {
    int nt = wv + t2 * 8;
    v8f acc[4];
#pragma unroll
    for (int m = 0; m < 4; ++m) acc[m] = zero8();
#pragma unroll
    for (int kt = 0; kt < 8; ++kt) {
      v16h b = b_frag(pw2, kt * 16 + nt, lane);
#pragma unroll
      for (int m = 0; m < 4; ++m) {
        v16h a = a_row(hid + (m * 16 + lm) * 256 + kt * 32, hs);  // ds_load_b128 x2
        acc[m] = wmma16(a, b, acc[m]);
      }
    }
    int col = nt * 16 + lm;
    float bv = b2[col];
#pragma unroll
    for (int m = 0; m < 4; ++m) {
      long ebase = (long)(tile0 + m) * 16;
      if (ebase + 16 <= E) {
#pragma unroll
        for (int r = 0; r < 8; ++r) {
          int dst = to_idx[ebase + r + 8 * hs];
          atomicAdd(&agg[(size_t)dst * 256 + col], acc[m][r] + bv);
        }
      } else {
#pragma unroll
        for (int r = 0; r < 8; ++r) {
          long e2 = ebase + r + 8 * hs;
          if (e2 < E) {
            int dst = to_idx[e2];
            atomicAdd(&agg[(size_t)dst * 256 + col], acc[m][r] + bv);
          }
        }
      }
    }
  }
}

// ---------------------------------------------------------------------------
// Residual node update: h += concat(h, agg)[N,384] @ W_upd[384,128] + b
// ---------------------------------------------------------------------------
__global__ __launch_bounds__(256) void k_node_update(
    const _Float16* __restrict__ h16, const float* __restrict__ agg,
    const _Float16* __restrict__ pwu, const float* __restrict__ bu,
    float* __restrict__ h, int N) {
  int tile = blockIdx.x;
  int lane = threadIdx.x & 31, nt = threadIdx.x >> 5;  // 8 waves = 8 n-tiles
  int hs = lane >> 4, lm = lane & 15;
  int arow = tile * 16 + lm; if (arow >= N) arow = N - 1;
  const _Float16* hr = h16 + (size_t)arow * 128;
  const float* ar = agg + (size_t)arow * 256;
  v8f acc = zero8();
#pragma unroll
  for (int kt = 0; kt < 12; ++kt) {
    int kb = kt * 32;
    int c0 = kb + 8 * hs, c1 = kb + 16 + 8 * hs;
    v8h x0 = (c0 < 128) ? *(const v8h*)(hr + c0) : cvt8f(ar + (c0 - 128));
    v8h x1 = (c1 < 128) ? *(const v8h*)(hr + c1) : cvt8f(ar + (c1 - 128));
    v16h a = combine8(x0, x1);
    v16h b = b_frag(pwu, kt * 8 + nt, lane);
    acc = wmma16(a, b, acc);
  }
  int col = nt * 16 + lm;
  float bv = bu[col];
  int nbase = tile * 16 + 8 * hs;
  if (tile * 16 + 16 <= N) {
#pragma unroll
    for (int r = 0; r < 8; ++r) {
      size_t ix = (size_t)(nbase + r) * 128 + col;
      h[ix] = h[ix] + acc[r] + bv;
    }
  } else {
#pragma unroll
    for (int r = 0; r < 8; ++r) {
      int node = nbase + r;
      if (node < N) {
        size_t ix = (size_t)node * 128 + col;
        h[ix] = h[ix] + acc[r] + bv;
      }
    }
  }
}

// ---------------------------------------------------------------------------
// Scatter node states into padded [2*B*MAX,128] f16 stacks.
// ---------------------------------------------------------------------------
__global__ void k_scatter(const float* __restrict__ h, const int* __restrict__ pos,
                          _Float16* __restrict__ flat16, long n /* N*128 */) {
  long i = (long)blockIdx.x * blockDim.x + threadIdx.x;
  if (i < n) {
    long node = i >> 7;
    int d = (int)(i & 127);
    int p = pos[node];
    flat16[(size_t)p * 128 + d] = (_Float16)h[i];
  }
}

// ---------------------------------------------------------------------------
// att_feat: t = mask * (relu(x@Wa1+ba1)@Wa2+ba2). 16 rows/block, LDS hidden.
// ---------------------------------------------------------------------------
__global__ __launch_bounds__(256) void k_att_feat(
    const _Float16* __restrict__ flat16,
    const _Float16* __restrict__ pw1, const float* __restrict__ b1,
    const _Float16* __restrict__ pw2, const float* __restrict__ b2,
    const int* __restrict__ qsz, const int* __restrict__ csz,
    _Float16* __restrict__ t16) {
  __shared__ __align__(16) _Float16 hid[16 * 128];
  int tile = blockIdx.x;  // 2048 tiles x 16 rows = 32768 rows
  int lane = threadIdx.x & 31, nt = threadIdx.x >> 5;
  int hs = lane >> 4, lm = lane & 15;
  int arow = tile * 16 + lm;
  const _Float16* fr = flat16 + (size_t)arow * 128;
  v8f acc = zero8();
#pragma unroll
  for (int kt = 0; kt < 4; ++kt) {
    v16h a = a_row(fr + kt * 32, hs);
    v16h b = b_frag(pw1, kt * 8 + nt, lane);
    acc = wmma16(a, b, acc);
  }
  int col = nt * 16 + lm;
  float bv = b1[col];
#pragma unroll
  for (int r = 0; r < 8; ++r) {
    float v = acc[r] + bv;
    v = v > 0.0f ? v : 0.0f;
    hid[(r + 8 * hs) * 128 + col] = (_Float16)v;
  }
  __syncthreads();
  v8f acc2 = zero8();
#pragma unroll
  for (int kt = 0; kt < 4; ++kt) {
    v16h a = a_row(hid + lm * 128 + kt * 32, hs);
    v16h b = b_frag(pw2, kt * 8 + nt, lane);
    acc2 = wmma16(a, b, acc2);
  }
  float bv2 = b2[col];
#pragma unroll
  for (int r = 0; r < 8; ++r) {
    int grow = tile * 16 + r + 8 * hs;
    int slot = grow & 63;
    int sz = (grow < 256 * 64) ? qsz[grow >> 6] : csz[(grow - 256 * 64) >> 6];
    float v = (slot < sz) ? (acc2[r] + bv2) : 0.0f;
    t16[(size_t)grow * 128 + col] = (_Float16)v;
  }
}

// ---------------------------------------------------------------------------
// Per graph-pair: logits (tq@tc^T / 0.1), masked dual softmax, hinge
// alignment scores, out[b] = min(q_score, c_score). One block per pair.
// tq/tc staged into LDS with global_load_async_to_lds_b128 (ASYNCcnt).
// ---------------------------------------------------------------------------
__global__ __launch_bounds__(256) void k_pair_scores(
    const _Float16* __restrict__ t16, const _Float16* __restrict__ flat16,
    const int* __restrict__ qsz, const int* __restrict__ csz,
    float* __restrict__ out) {
  __shared__ __align__(16) _Float16 tq[64 * 128];
  __shared__ __align__(16) _Float16 tc[64 * 128];
  __shared__ float lg[64 * 64];
  __shared__ float rmax[64], rsum[64], cmax2[64], csum2[64];
  __shared__ float red[256];
  __shared__ float qneg;
  int b = blockIdx.x, t = threadIdx.x;
  const _Float16* gq = t16 + (size_t)b * 8192;
  const _Float16* gc = t16 + (size_t)(16384 + b * 64) * 128;
  // Async cooperative copy: 16 KB per tile, 16B chunks per lane.
  unsigned ldq = (unsigned)(size_t)(void*)tq;  // LDS offset (low 32 bits of generic addr)
  unsigned ldc = (unsigned)(size_t)(void*)tc;
#pragma unroll
  for (int i = 0; i < 4; ++i) {
    int chunk = t + i * 256;  // 1024 chunks of 16B
    unsigned dq = ldq + chunk * 16;
    unsigned dc = ldc + chunk * 16;
    unsigned long long sqa = (unsigned long long)(size_t)gq + (size_t)chunk * 16;
    unsigned long long sca = (unsigned long long)(size_t)gc + (size_t)chunk * 16;
    asm volatile("global_load_async_to_lds_b128 %0, %1, off" :: "v"(dq), "v"(sqa) : "memory");
    asm volatile("global_load_async_to_lds_b128 %0, %1, off" :: "v"(dc), "v"(sca) : "memory");
  }
  asm volatile("s_wait_asynccnt 0x0" ::: "memory");
  int nq = qsz[b], nc = csz[b];
  __syncthreads();
  // logits
  for (int i = t; i < 4096; i += 256) {
    int q = i >> 6, c = i & 63;
    const _Float16* a = tq + q * 128;
    const _Float16* bb = tc + c * 128;
    float d = 0.0f;
    for (int k = 0; k < 128; ++k) d += (float)a[k] * (float)bb[k];
    lg[i] = (q < nq && c < nc) ? d * 10.0f : -1e9f;  // 1/TEMP = 10
  }
  __syncthreads();
  // row / column softmax stats
  if (t < 64) {
    float m = -3.0e38f;
    for (int c = 0; c < 64; ++c) m = fmaxf(m, lg[t * 64 + c]);
    float s = 0.0f;
    for (int c = 0; c < 64; ++c) s += __expf(lg[t * 64 + c] - m);
    rmax[t] = m; rsum[t] = s;
  } else if (t < 128) {
    int c = t - 64;
    float m = -3.0e38f;
    for (int q = 0; q < 64; ++q) m = fmaxf(m, lg[q * 64 + c]);
    float s = 0.0f;
    for (int q = 0; q < 64; ++q) s += __expf(lg[q * 64 + c] - m);
    cmax2[c] = m; csum2[c] = s;
  }
  __syncthreads();
  const _Float16* sq = flat16 + (size_t)b * 8192;
  const _Float16* sc = flat16 + (size_t)(16384 + b * 64) * 128;
  // q -> c alignment (4 threads per q row, 32 d-cols each)
  float part = 0.0f;
  {
    int q = t >> 2, d0 = (t & 3) * 32;
    if (q < nq) {
      float w[32];
#pragma unroll
      for (int i = 0; i < 32; ++i) w[i] = 0.0f;
      float rm = rmax[q], rs = rsum[q];
      for (int c = 0; c < nc; ++c) {
        float p = __expf(lg[q * 64 + c] - rm) / rs;
        const _Float16* scr = sc + (size_t)c * 128 + d0;
#pragma unroll
        for (int i = 0; i < 32; ++i) w[i] += p * (float)scr[i];
      }
      const _Float16* sqr = sq + (size_t)q * 128 + d0;
#pragma unroll
      for (int i = 0; i < 32; ++i) {
        float df = (float)sqr[i] - w[i];
        part += df > 0.0f ? df : 0.0f;
      }
    }
  }
  red[t] = part;
  __syncthreads();
  for (int s = 128; s > 0; s >>= 1) { if (t < s) red[t] += red[t + s]; __syncthreads(); }
  if (t == 0) qneg = -red[0];
  __syncthreads();
  // c -> q alignment
  part = 0.0f;
  {
    int c = t >> 2, d0 = (t & 3) * 32;
    if (c < nc) {
      float w[32];
#pragma unroll
      for (int i = 0; i < 32; ++i) w[i] = 0.0f;
      float cm = cmax2[c], cs = csum2[c];
      for (int q = 0; q < nq; ++q) {
        float p = __expf(lg[q * 64 + c] - cm) / cs;
        const _Float16* sqr = sq + (size_t)q * 128 + d0;
#pragma unroll
        for (int i = 0; i < 32; ++i) w[i] += p * (float)sqr[i];
      }
      const _Float16* scr = sc + (size_t)c * 128 + d0;
#pragma unroll
      for (int i = 0; i < 32; ++i) {
        float df = (float)scr[i] - w[i];
        part += df > 0.0f ? df : 0.0f;
      }
    }
  }
  red[t] = part;
  __syncthreads();
  for (int s = 128; s > 0; s >>= 1) { if (t < s) red[t] += red[t + s]; __syncthreads(); }
  if (t == 0) out[b] = fminf(qneg, -red[0]);
}

// ---------------------------------------------------------------------------
// Host driver
// ---------------------------------------------------------------------------
extern "C" void kernel_launch(void* const* d_in, const int* in_sizes, int n_in,
                              void* d_out, int out_size, void* d_ws, size_t ws_size,
                              hipStream_t stream) {
  (void)n_in; (void)out_size; (void)ws_size;
  const float* nf = (const float*)d_in[0];
  const float* ef = (const float*)d_in[1];
  const int* from_idx = (const int*)d_in[2];
  const int* to_idx = (const int*)d_in[3];
  const int* node_to_pos = (const int*)d_in[4];
  const int* qsz = (const int*)d_in[5];
  const int* csz = (const int*)d_in[6];
  const float* W_node = (const float*)d_in[7];  const float* b_node = (const float*)d_in[8];
  const float* W_edge = (const float*)d_in[9];  const float* b_edge = (const float*)d_in[10];
  const float* W_msg1 = (const float*)d_in[11]; const float* b_msg1 = (const float*)d_in[12];
  const float* W_msg2 = (const float*)d_in[13]; const float* b_msg2 = (const float*)d_in[14];
  const float* W_upd  = (const float*)d_in[15]; const float* b_upd  = (const float*)d_in[16];
  const float* Wa1 = (const float*)d_in[17]; const float* ba1 = (const float*)d_in[18];
  const float* Wa2 = (const float*)d_in[19]; const float* ba2 = (const float*)d_in[20];

  long N = (long)in_sizes[0] / 32;
  long E = (long)in_sizes[1] / 16;

  char* base = (char*)d_ws;
  size_t off = 0;
  auto alloc = [&](size_t bytes) -> void* {
    void* p = base + off;
    off += (bytes + 255) & ~(size_t)255;
    return p;
  };
  float*    h      = (float*)alloc((size_t)N * 128 * 4);
  _Float16* h16    = (_Float16*)alloc((size_t)N * 128 * 2);
  _Float16* nf16   = (_Float16*)alloc((size_t)N * 32 * 2);
  _Float16* ef16   = (_Float16*)alloc((size_t)E * 32 * 2);
  _Float16* e16    = (_Float16*)alloc((size_t)E * 64 * 2);
  float*    agg    = (float*)alloc((size_t)N * 256 * 4);
  _Float16* flat16 = (_Float16*)alloc((size_t)2 * 256 * 64 * 128 * 2);
  _Float16* t16    = (_Float16*)alloc((size_t)2 * 256 * 64 * 128 * 2);
  _Float16* pwN  = (_Float16*)alloc((size_t)32 * 128 * 2);
  _Float16* pwE  = (_Float16*)alloc((size_t)32 * 64 * 2);
  _Float16* pw1  = (_Float16*)alloc((size_t)320 * 256 * 2);
  _Float16* pw2  = (_Float16*)alloc((size_t)256 * 256 * 2);
  _Float16* pwU  = (_Float16*)alloc((size_t)384 * 128 * 2);
  _Float16* pwA1 = (_Float16*)alloc((size_t)128 * 128 * 2);
  _Float16* pwA2 = (_Float16*)alloc((size_t)128 * 128 * 2);

  // --- pack weights into WMMA B-fragment order ---
  k_pack_bfrag<<<(32 / 32) * (128 / 16), 32, 0, stream>>>(W_node, pwN, 32, 32, 128);
  k_pack_bfrag<<<(32 / 32) * (64 / 16), 32, 0, stream>>>(W_edge, pwE, 16, 32, 64);
  k_pack_bfrag<<<(320 / 32) * (256 / 16), 32, 0, stream>>>(W_msg1, pw1, 320, 320, 256);
  k_pack_bfrag<<<(256 / 32) * (256 / 16), 32, 0, stream>>>(W_msg2, pw2, 256, 256, 256);
  k_pack_bfrag<<<(384 / 32) * (128 / 16), 32, 0, stream>>>(W_upd, pwU, 384, 384, 128);
  k_pack_bfrag<<<(128 / 32) * (128 / 16), 32, 0, stream>>>(Wa1, pwA1, 128, 128, 128);
  k_pack_bfrag<<<(128 / 32) * (128 / 16), 32, 0, stream>>>(Wa2, pwA2, 128, 128, 128);

  // --- f16 staging of inputs ---
  long nNF = N * 32;
  k_cvt_f16<<<(int)((nNF + 255) / 256), 256, 0, stream>>>(nf, nf16, nNF);
  long nEF = E * 32;
  k_cvt_pad32<<<(int)((nEF + 255) / 256), 256, 0, stream>>>(ef, ef16, E);

  int NT_N = (int)((N + 15) / 16);
  int NT_E = (int)((E + 15) / 16);
  int EB4  = (int)((E + 63) / 64);

  // --- encoders ---
  k_encode_node<<<NT_N, 256, 0, stream>>>(nf16, pwN, b_node, h, h16, (int)N);
  k_encode_edge<<<NT_E, 128, 0, stream>>>(ef16, pwE, b_edge, e16, E);

  // --- 3 propagation steps ---
  for (int step = 0; step < 3; ++step) {
    long nAgg = N * 256;
    k_zero_u32<<<(int)((nAgg + 255) / 256), 256, 0, stream>>>((unsigned int*)agg, nAgg);
    k_edge_msg<<<EB4, 256, 0, stream>>>(h16, e16, from_idx, to_idx,
                                        pw1, b_msg1, pw2, b_msg2, agg, E);
    k_node_update<<<NT_N, 256, 0, stream>>>(h16, agg, pwU, b_upd, h, (int)N);
    long nH = N * 128;
    k_cvt_f16<<<(int)((nH + 255) / 256), 256, 0, stream>>>(h, h16, nH);
  }

  // --- scatter into padded stacks (zero first: padded slots must be 0) ---
  long nFlatU32 = (long)2 * 256 * 64 * 128 / 2;
  k_zero_u32<<<(int)((nFlatU32 + 255) / 256), 256, 0, stream>>>((unsigned int*)flat16, nFlatU32);
  long nH = N * 128;
  k_scatter<<<(int)((nH + 255) / 256), 256, 0, stream>>>(h, node_to_pos, flat16, nH);

  // --- attention features + pair scores ---
  k_att_feat<<<2048, 256, 0, stream>>>(flat16, pwA1, ba1, pwA2, ba2, qsz, csz, t16);
  k_pair_scores<<<256, 256, 0, stream>>>(t16, flat16, qsz, csz, (float*)d_out);
}
